// MultiHeadSelfAttention_52372831207733
// MI455X (gfx1250) — matrix operand order repaired and checked by reading, not compiled, not run on hardware
//
#include <hip/hip_runtime.h>

// ---------------------------------------------------------------------------
// MHA forward for MI455X (gfx1250, wave32, WMMA bf16 path).
// Pipeline: cast f32->bf16 -> QKV GEMMs (wmma bf16, async-LDS double-buffered)
// -> flash attention (wmma bf16, online softmax) -> output projection (f32).
// ---------------------------------------------------------------------------

typedef __attribute__((ext_vector_type(16))) __bf16 v16bf;
typedef __attribute__((ext_vector_type(8)))  __bf16 v8bf;
typedef __attribute__((ext_vector_type(8)))  float  v8f;

#define S_LEN 2048
#define BATCH 4
#define DM    1024
#define NH    16
#define DKH   64
#define MROWS (BATCH * S_LEN)        // 8192
#define NEG_INF_F (-1000000000.0f)

// ---------------------------------------------------------------------------
// Async global -> LDS copy, 16 bytes per lane (GLOBAL_LOAD_ASYNC_TO_LDS_B128,
// GV mode: 64-bit global address in VGPR pair, LDS byte offset in VGPR).
// Tracked by ASYNCcnt; retire with s_wait_asynccnt.
// ---------------------------------------------------------------------------
__device__ __forceinline__ void async_copy_b128(const __bf16* gsrc, __bf16* ldst) {
  const unsigned loff = (unsigned)(unsigned long long)(uintptr_t)ldst;  // LDS byte addr = flat[31:0]
  asm volatile("global_load_async_to_lds_b128 %0, %1, off"
               :: "v"(loff), "v"(gsrc)
               : "memory");
}

// ---------------------------------------------------------------------------
// f32 -> bf16 cast (vectorized x4)
// ---------------------------------------------------------------------------
__global__ __launch_bounds__(256) void cast_f32_bf16(
    const float* __restrict__ in, __bf16* __restrict__ out, int n) {
  int i = (blockIdx.x * blockDim.x + threadIdx.x) * 4;
  if (i + 3 < n) {
    float4 f = *(const float4*)(in + i);
    out[i + 0] = (__bf16)f.x;
    out[i + 1] = (__bf16)f.y;
    out[i + 2] = (__bf16)f.z;
    out[i + 3] = (__bf16)f.w;
  }
}

// ---------------------------------------------------------------------------
// Tiled GEMM: C[m][n] = sum_k A[m][k] * W[n][k] + bias[n]
// A: [M][K] bf16 row-major; W: [N][K] bf16 row-major (weight as stored, [e][d])
// Block tile 128x128, 8 waves (2x4), wave tile 64x32, K-step 32.
// Double-buffered LDS filled with async global->LDS B128 copies.
// ---------------------------------------------------------------------------
template <bool OUT_F32>
__global__ __launch_bounds__(256) void gemm_xw(
    const __bf16* __restrict__ A, const __bf16* __restrict__ W,
    const float* __restrict__ bias, void* __restrict__ Cout,
    int Mdim, int Ndim, int Kdim) {
  constexpr int LDA = 40;  // 32 + 8 bf16 pad: 80B row stride, keeps 16B chunks aligned
  __shared__ __bf16 lds_a[2][128 * LDA];
  __shared__ __bf16 lds_b[2][128 * LDA];

  const int tid  = threadIdx.x;
  const int lane = tid & 31;
  const int wave = tid >> 5;
  const int hlf  = lane >> 4;   // lane half (wave32)
  const int l16  = lane & 15;
  const int wm   = (wave >> 2) * 64;  // wave row offset in block tile
  const int wn   = (wave & 3) * 32;   // wave col offset in block tile
  const int row0 = blockIdx.y * 128;
  const int col0 = blockIdx.x * 128;

  // Per-thread staging assignment: chunks tid and tid+256 of 512 16B chunks.
  const int rs0  = tid >> 2;            // staging row 0..63
  const int rs1  = rs0 + 64;            // staging row 64..127
  const int offs = (tid & 3) * 8;       // bf16 offset within 32-wide K slice

  v8f acc[4][2] = {};

  const int NIT = Kdim / 32;
  // Issue one K-slice (A 128x32 + W 128x32) into LDS buffer `buf`:
  // 4 async B128 per thread.
  auto issue = [&](int buf, int k0) {
    async_copy_b128(&A[(size_t)(row0 + rs0) * Kdim + k0 + offs],
                    &lds_a[buf][rs0 * LDA + offs]);
    async_copy_b128(&A[(size_t)(row0 + rs1) * Kdim + k0 + offs],
                    &lds_a[buf][rs1 * LDA + offs]);
    async_copy_b128(&W[(size_t)(col0 + rs0) * Kdim + k0 + offs],
                    &lds_b[buf][rs0 * LDA + offs]);
    async_copy_b128(&W[(size_t)(col0 + rs1) * Kdim + k0 + offs],
                    &lds_b[buf][rs1 * LDA + offs]);
  };

  issue(0, 0);
  for (int it = 0; it < NIT; ++it) {
    const int cur = it & 1;
    if (it + 1 < NIT) {
      issue(cur ^ 1, (it + 1) * 32);                     // prefetch next slice
      asm volatile("s_wait_asynccnt 4" ::: "memory");    // retire current slice only
    } else {
      asm volatile("s_wait_asynccnt 0" ::: "memory");
    }
    __syncthreads();  // all waves' current slice landed in LDS

    // A fragments: lane half selects K {0..7,16..23} vs {8..15,24..31}
    v16bf afr[4];
#pragma unroll
    for (int mt = 0; mt < 4; ++mt) {
      const int m = wm + mt * 16 + l16;
      union { v16bf v; v8bf h[2]; } u;
      u.h[0] = *(const v8bf*)&lds_a[cur][m * LDA + hlf * 8];
      u.h[1] = *(const v8bf*)&lds_a[cur][m * LDA + 16 + hlf * 8];
      afr[mt] = u.v;
    }
    // B fragments: lane half selects K {0..7,16..23} vs {8..15,24..31}
    v16bf bfr[2];
#pragma unroll
    for (int nt = 0; nt < 2; ++nt) {
      const int n = wn + nt * 16 + l16;
      union { v16bf v; v8bf h[2]; } u;
      u.h[0] = *(const v8bf*)&lds_b[cur][n * LDA + hlf * 8];
      u.h[1] = *(const v8bf*)&lds_b[cur][n * LDA + 16 + hlf * 8];
      bfr[nt] = u.v;
    }
#pragma unroll
    for (int mt = 0; mt < 4; ++mt)
#pragma unroll
      for (int nt = 0; nt < 2; ++nt)
        acc[mt][nt] = __builtin_amdgcn_wmma_f32_16x16x32_bf16(
            false, afr[mt], false, bfr[nt], (short)0, acc[mt][nt], false, false);

    __syncthreads();  // protect `cur` buffer before next iteration re-issues into it
  }

  // Epilogue: C layout is M = r + 8*half, N = l16 within each 16x16 tile.
#pragma unroll
  for (int nt = 0; nt < 2; ++nt) {
    const int n  = col0 + wn + nt * 16 + l16;
    const float bv = bias[n];
#pragma unroll
    for (int mt = 0; mt < 4; ++mt) {
#pragma unroll
      for (int r = 0; r < 8; ++r) {
        const int m = row0 + wm + mt * 16 + r + hlf * 8;
        const float v = acc[mt][nt][r] + bv;
        if (OUT_F32)
          ((float*)Cout)[(size_t)m * Ndim + n] = v;
        else
          ((__bf16*)Cout)[(size_t)m * Ndim + n] = (__bf16)v;
      }
    }
  }
}

// ---------------------------------------------------------------------------
// Flash attention per (b, h): 128 queries/block (16 per wave), key tiles of
// 64 staged in LDS (V transposed on store). Online softmax; pad mask applied
// as NEG_INF replacement on key columns. Output bf16 [B*S][D].
// ---------------------------------------------------------------------------
__global__ __launch_bounds__(256) void flash_attn(
    const __bf16* __restrict__ Q, const __bf16* __restrict__ K,
    const __bf16* __restrict__ V, const int* __restrict__ pad,
    __bf16* __restrict__ O) {
  constexpr int LDK = 72;  // 64 + 8 bf16 pad
  __shared__ __bf16   lds_k [64 * LDK];
  __shared__ __bf16   lds_vt[64 * LDK];
  __shared__ __bf16   lds_p [8 * 16 * LDK];  // per-wave P scratch
  __shared__ unsigned lds_mask[64];

  const int tid  = threadIdx.x;
  const int lane = tid & 31;
  const int wave = tid >> 5;
  const int hlf  = lane >> 4;
  const int l16  = lane & 15;
  const int q0   = blockIdx.x * 128;
  const int h    = blockIdx.y;
  const int b    = blockIdx.z;
  const size_t rowbase = (size_t)b * S_LEN;
  const int dcol = h * DKH;

  // Q fragments for this wave's 16 rows (d_k = 64 -> two K=32 chunks), kept live.
  v16bf qa[2];
  {
    const int m = q0 + wave * 16 + l16;
    const __bf16* qrow = Q + (rowbase + m) * DM + dcol;
#pragma unroll
    for (int c = 0; c < 2; ++c) {
      union { v16bf v; v8bf hh[2]; } u;
      u.hh[0] = *(const v8bf*)&qrow[c * 32 + hlf * 8];
      u.hh[1] = *(const v8bf*)&qrow[c * 32 + 16 + hlf * 8];
      qa[c] = u.v;
    }
  }

  v8f o[4] = {};
  float mst[8], lst[8];
#pragma unroll
  for (int r = 0; r < 8; ++r) { mst[r] = -__builtin_inff(); lst[r] = 0.f; }

  __bf16* pbuf = &lds_p[wave * 16 * LDK];

  for (int kt0 = 0; kt0 < S_LEN; kt0 += 64) {
    __syncthreads();
    // Stage K (row-major) and V (transposed) tiles: 64 keys x 64 d_k.
    int c = tid;
#pragma unroll
    for (int rep = 0; rep < 2; ++rep, c += 256) {
      const int key = c >> 3;
      const int off = (c & 7) * 8;
      const size_t gr = (rowbase + kt0 + key) * DM + dcol + off;
      *(v8bf*)&lds_k[key * LDK + off] = *(const v8bf*)&K[gr];
      v8bf vv = *(const v8bf*)&V[gr];
#pragma unroll
      for (int j = 0; j < 8; ++j) lds_vt[(off + j) * LDK + key] = vv[j];
    }
    if (tid < 64) lds_mask[tid] = (pad[b * S_LEN + kt0 + tid] != 0) ? 1u : 0u;
    __syncthreads();

    // Scores: 4 key sub-tiles of 16, each 2 WMMAs over d_k=64.
    v8f st[4];
#pragma unroll
    for (int t = 0; t < 4; ++t) {
      const int key = t * 16 + l16;
      v8f accs = {};
#pragma unroll
      for (int cc = 0; cc < 2; ++cc) {
        union { v16bf v; v8bf hh[2]; } u;
        u.hh[0] = *(const v8bf*)&lds_k[key * LDK + cc * 32 + hlf * 8];
        u.hh[1] = *(const v8bf*)&lds_k[key * LDK + cc * 32 + 16 + hlf * 8];
        accs = __builtin_amdgcn_wmma_f32_16x16x32_bf16(
            false, qa[cc], false, u.v, (short)0, accs, false, false);
      }
      st[t] = accs;
    }
    // Scale by 1/sqrt(d_k)=0.125 then replace masked key columns with NEG_INF.
#pragma unroll
    for (int t = 0; t < 4; ++t) {
      const bool msk = lds_mask[t * 16 + l16] != 0;
#pragma unroll
      for (int r = 0; r < 8; ++r) {
        const float v = st[t][r] * 0.125f;
        st[t][r] = msk ? NEG_INF_F : v;
      }
    }
    // Row max (reduce over 16 lanes within each half — rows live per half).
    float cmax[8];
#pragma unroll
    for (int r = 0; r < 8; ++r) {
      float v = fmaxf(fmaxf(st[0][r], st[1][r]), fmaxf(st[2][r], st[3][r]));
#pragma unroll
      for (int off = 1; off < 16; off <<= 1) v = fmaxf(v, __shfl_xor(v, off, 32));
      cmax[r] = v;
    }
    float scl[8];
#pragma unroll
    for (int r = 0; r < 8; ++r) {
      const float mn = fmaxf(mst[r], cmax[r]);
      scl[r] = __expf(mst[r] - mn);
      mst[r] = mn;
    }
    // exp, store P (bf16) into per-wave LDS scratch, accumulate partial sums.
    float psum[8] = {};
#pragma unroll
    for (int t = 0; t < 4; ++t) {
#pragma unroll
      for (int r = 0; r < 8; ++r) {
        const float p = __expf(st[t][r] - mst[r]);
        psum[r] += p;
        pbuf[(r + hlf * 8) * LDK + t * 16 + l16] = (__bf16)p;
      }
    }
#pragma unroll
    for (int r = 0; r < 8; ++r) {
      float v = psum[r];
#pragma unroll
      for (int off = 1; off < 16; off <<= 1) v += __shfl_xor(v, off, 32);
      lst[r] = lst[r] * scl[r] + v;
    }
    // Rescale running O, then O += P * V (A-frags re-loaded from LDS scratch).
#pragma unroll
    for (int nt = 0; nt < 4; ++nt)
#pragma unroll
      for (int r = 0; r < 8; ++r) o[nt][r] *= scl[r];

#pragma unroll
    for (int cc = 0; cc < 2; ++cc) {
      union { v16bf v; v8bf hh[2]; } pa;
      pa.hh[0] = *(const v8bf*)&pbuf[l16 * LDK + cc * 32 + hlf * 8];
      pa.hh[1] = *(const v8bf*)&pbuf[l16 * LDK + cc * 32 + 16 + hlf * 8];
#pragma unroll
      for (int nt = 0; nt < 4; ++nt) {
        const int n = nt * 16 + l16;
        union { v16bf v; v8bf hh[2]; } u;
        u.hh[0] = *(const v8bf*)&lds_vt[n * LDK + cc * 32 + hlf * 8];
        u.hh[1] = *(const v8bf*)&lds_vt[n * LDK + cc * 32 + 16 + hlf * 8];
        o[nt] = __builtin_amdgcn_wmma_f32_16x16x32_bf16(
            false, pa.v, false, u.v, (short)0, o[nt], false, false);
      }
    }
  }

  // Epilogue: divide by row sums, write bf16 [B,S,H,dk] (== [B*S][D]).
#pragma unroll
  for (int nt = 0; nt < 4; ++nt) {
#pragma unroll
    for (int r = 0; r < 8; ++r) {
      const int m = q0 + wave * 16 + r + hlf * 8;
      const float v = o[nt][r] / lst[r];
      O[(rowbase + m) * DM + dcol + nt * 16 + l16] = (__bf16)v;
    }
  }
}

// ---------------------------------------------------------------------------
// Launcher. Workspace layout (bytes):
//   xb      @ 0         16 MiB   (x bf16)
//   wqb..wob@ 16 MiB    4 x 2 MiB
//   Qb      @ 24 MiB    16 MiB
//   Kb      @ 40 MiB    16 MiB
//   Vb      @ 56 MiB    16 MiB
//   attnb   @ 72 MiB    16 MiB    -> total 88 MiB
// ---------------------------------------------------------------------------
extern "C" void kernel_launch(void* const* d_in, const int* in_sizes, int n_in,
                              void* d_out, int out_size, void* d_ws, size_t ws_size,
                              hipStream_t stream) {
  (void)in_sizes; (void)n_in; (void)out_size; (void)ws_size;
  const float* x  = (const float*)d_in[0];
  const int*   pad= (const int*)  d_in[1];
  const float* wq = (const float*)d_in[2];
  const float* bq = (const float*)d_in[3];
  const float* wk = (const float*)d_in[4];
  const float* bk = (const float*)d_in[5];
  const float* wv = (const float*)d_in[6];
  const float* bv = (const float*)d_in[7];
  const float* wo = (const float*)d_in[8];
  const float* bo = (const float*)d_in[9];

  char* ws = (char*)d_ws;
  const size_t MB = 1024 * 1024;
  __bf16* xb    = (__bf16*)(ws + 0);
  __bf16* wqb   = (__bf16*)(ws + 16 * MB);
  __bf16* wkb   = (__bf16*)(ws + 18 * MB);
  __bf16* wvb   = (__bf16*)(ws + 20 * MB);
  __bf16* wob   = (__bf16*)(ws + 22 * MB);
  __bf16* Qb    = (__bf16*)(ws + 24 * MB);
  __bf16* Kb    = (__bf16*)(ws + 40 * MB);
  __bf16* Vb    = (__bf16*)(ws + 56 * MB);
  __bf16* attnb = (__bf16*)(ws + 72 * MB);

  const int nX = MROWS * DM;   // 8,388,608
  const int nW = DM * DM;      // 1,048,576
  cast_f32_bf16<<<nX / 4 / 256, 256, 0, stream>>>(x,  xb,  nX);
  cast_f32_bf16<<<nW / 4 / 256, 256, 0, stream>>>(wq, wqb, nW);
  cast_f32_bf16<<<nW / 4 / 256, 256, 0, stream>>>(wk, wkb, nW);
  cast_f32_bf16<<<nW / 4 / 256, 256, 0, stream>>>(wv, wvb, nW);
  cast_f32_bf16<<<nW / 4 / 256, 256, 0, stream>>>(wo, wob, nW);

  dim3 gg(DM / 128, MROWS / 128);  // (8, 64)
  gemm_xw<false><<<gg, 256, 0, stream>>>(xb, wqb, bq, Qb, MROWS, DM, DM);
  gemm_xw<false><<<gg, 256, 0, stream>>>(xb, wkb, bk, Kb, MROWS, DM, DM);
  gemm_xw<false><<<gg, 256, 0, stream>>>(xb, wvb, bv, Vb, MROWS, DM, DM);

  dim3 ga(S_LEN / 128, NH, BATCH);  // (16, 16, 4)
  flash_attn<<<ga, 256, 0, stream>>>(Qb, Kb, Vb, pad, attnb);

  gemm_xw<true><<<gg, 256, 0, stream>>>(attnb, wob, bo, (float*)d_out, MROWS, DM, DM);
}
